// AgreemFlat_33964601377532
// MI455X (gfx1250) — matrix operand order repaired and checked by reading, not compiled, not run on hardware
//
#include <hip/hip_runtime.h>

// ---------------------------------------------------------------------------
// Problem constants (from reference)
// ---------------------------------------------------------------------------
constexpr int B        = 4096;
constexpr int N        = 64;
constexpr int D_SIM    = 768;
constexpr int NLI_DIM  = 768;
constexpr int KK       = 5;
constexpr int FAN1     = (KK + 1) * NLI_DIM;   // 4608
constexpr int H1       = 1024;
constexpr int NCLS     = 4;

// GEMM1 tiling: block tile 256x64, 8 waves, each wave owns a 32x64 strip.
constexpr int BM = 256;
constexpr int BN = 64;
constexpr int KC = 32;                 // K chunk == WMMA K depth
constexpr int NCHUNK = FAN1 / KC;      // 144

typedef __attribute__((ext_vector_type(16))) __bf16 v16bf;
typedef __attribute__((ext_vector_type(8)))  __bf16 v8bf;
typedef __attribute__((ext_vector_type(8)))  float  v8f;
typedef __attribute__((ext_vector_type(4)))  int    v4i;

#ifndef __has_builtin
#define __has_builtin(x) 0
#endif

#if __has_builtin(__builtin_amdgcn_global_load_async_to_lds_b128)
#define HAVE_ASYNC_LDS 1
#else
#define HAVE_ASYNC_LDS 0
#endif

#if HAVE_ASYNC_LDS
#if __has_builtin(__builtin_amdgcn_s_wait_asynccnt)
#define WAIT_ASYNC(n) __builtin_amdgcn_s_wait_asynccnt(n)
#else
#define WAIT_ASYNC(n) asm volatile("s_wait_asynccnt %0" :: "i"(n) : "memory")
#endif
#else
#define WAIT_ASYNC(n) ((void)0)
#endif

// 16-byte global -> LDS copy; async (no VGPR round trip, ASYNCcnt-tracked).
// Builtin prototype (from hipcc diagnostic): (v4i AS1* src, v4i AS3* dst, Ii, Ii)
__device__ __forceinline__ void copy16_g2l(void* lds, const void* gsrc) {
#if HAVE_ASYNC_LDS
    __builtin_amdgcn_global_load_async_to_lds_b128(
        (__attribute__((address_space(1))) v4i*)((void*)gsrc),
        (__attribute__((address_space(3))) v4i*)(lds),
        0, 0);
#else
    *reinterpret_cast<uint4*>(lds) = *reinterpret_cast<const uint4*>(gsrc);
#endif
}

__device__ __forceinline__ unsigned short f2bf(float f) {
    unsigned u = __float_as_uint(f);
    u += 0x7FFFu + ((u >> 16) & 1u);
    return (unsigned short)(u >> 16);
}

// ---------------------------------------------------------------------------
// Kernel 1: sims[b,n] = <sim_stance[b], sim_body[b,n]>, then top-5 indices.
// Bandwidth-bound (805 MB stream): float4 loads, wave32 xor-shuffle reduce.
// ---------------------------------------------------------------------------
__global__ __launch_bounds__(256) void sims_topk_kernel(
    const float* __restrict__ stance,   // [B, D_SIM]
    const float* __restrict__ body,     // [B, N, D_SIM]
    int* __restrict__ topidx)           // [B, KK]
{
    __shared__ __align__(16) float s_stance[D_SIM];
    __shared__ float s_sims[N];

    const int b = blockIdx.x;
    for (int i = threadIdx.x; i < D_SIM; i += 256)
        s_stance[i] = stance[(size_t)b * D_SIM + i];
    __syncthreads();

    const int wave = threadIdx.x >> 5;
    const int lane = threadIdx.x & 31;
    const float4* st4 = reinterpret_cast<const float4*>(s_stance);
    const float* bodyB = body + (size_t)b * N * D_SIM;

    for (int n = wave; n < N; n += 8) {
        const float4* row4 = reinterpret_cast<const float4*>(bodyB + (size_t)n * D_SIM);
        float acc = 0.f;
        #pragma unroll
        for (int i = lane; i < D_SIM / 4; i += 32) {
            float4 x = row4[i];
            float4 y = st4[i];
            acc += x.x * y.x + x.y * y.y + x.z * y.z + x.w * y.w;
        }
        #pragma unroll
        for (int off = 16; off >= 1; off >>= 1)
            acc += __shfl_xor(acc, off, 32);
        if (lane == 0) s_sims[n] = acc;
    }
    __syncthreads();

    if (threadIdx.x == 0) {
        #pragma unroll
        for (int k = 0; k < KK; ++k) {
            int   best = 0;
            float bv   = s_sims[0];
            for (int n = 1; n < N; ++n) {
                float v = s_sims[n];
                if (v > bv) { bv = v; best = n; }   // first-max wins: jax tie order
            }
            topidx[b * KK + k] = best;
            s_sims[best] = -__builtin_inff();
        }
    }
}

// ---------------------------------------------------------------------------
// Kernel 2: xx_bf16[b] = concat(nli_stance[b], gathered top-5 nli_body rows)
// ---------------------------------------------------------------------------
__global__ __launch_bounds__(256) void build_xx_kernel(
    const float* __restrict__ nli_stance,
    const float* __restrict__ nli_body,
    const int*   __restrict__ topidx,
    unsigned short* __restrict__ xx)        // [B, FAN1] bf16
{
    const int b = blockIdx.x;
    unsigned short* dst = xx + (size_t)b * FAN1;

    for (int i = threadIdx.x; i < NLI_DIM; i += 256)
        dst[i] = f2bf(nli_stance[(size_t)b * NLI_DIM + i]);

    #pragma unroll
    for (int k = 0; k < KK; ++k) {
        const int idx = topidx[b * KK + k];
        const float* src = nli_body + ((size_t)b * N + idx) * NLI_DIM;
        unsigned short* d = dst + (size_t)(k + 1) * NLI_DIM;
        for (int i = threadIdx.x; i < NLI_DIM; i += 256)
            d[i] = f2bf(src[i]);
    }
}

// ---------------------------------------------------------------------------
// Kernel 3: W1 fp32 -> bf16 (9.4 MB; L2-resident for GEMM1 reuse x4096)
// ---------------------------------------------------------------------------
__global__ __launch_bounds__(256) void cvt_w1_kernel(
    const float* __restrict__ w1, unsigned short* __restrict__ w1b, int total)
{
    for (int i = blockIdx.x * 256 + threadIdx.x; i < total; i += gridDim.x * 256)
        w1b[i] = f2bf(w1[i]);
}

// ---------------------------------------------------------------------------
// Kernel 4: H = relu(xx @ W1^T + b1), v_wmma_f32_16x16x32_bf16.
// Double-buffered LDS fed by GLOBAL_LOAD_ASYNC_TO_LDS_B128 (ASYNCcnt),
// per-wave 32x64 strip: 2 A frags x 4 B frags -> 8 WMMA per chunk.
// Named accumulators so regalloc can keep WMMA C/D in place (no v_mov /
// hazard-NOP churn at unroll seams).
// ---------------------------------------------------------------------------
__global__ __launch_bounds__(256) void gemm1_kernel(
    const unsigned short* __restrict__ A,    // xx  [B, FAN1] bf16
    const unsigned short* __restrict__ Bw,   // W1  [H1, FAN1] bf16
    const float* __restrict__ bias,          // [H1]
    float* __restrict__ H)                   // [B, H1] fp32
{
    __shared__ __align__(16) unsigned short sA[2][BM][KC];  // 32 KB
    __shared__ __align__(16) unsigned short sB[2][BN][KC];  //  8 KB

    const int tile_m = blockIdx.x * BM;
    const int tile_n = blockIdx.y * BN;
    const int tid  = threadIdx.x;
    const int wave = tid >> 5;
    const int lane = tid & 31;
    const int wrow = wave * 32;            // wave's 32-row strip in block tile
    const int r16  = lane & 15;
    const int hi   = lane >> 4;            // 0 or 1
    const int kA   = hi * 8;               // A frag K base within chunk
    const int kB   = hi * 8;               // B frag K base within chunk

    const v8f vzero = {0.f,0.f,0.f,0.f,0.f,0.f,0.f,0.f};
    v8f c00 = vzero, c01 = vzero, c02 = vzero, c03 = vzero;
    v8f c10 = vzero, c11 = vzero, c12 = vzero, c13 = vzero;

    // Precompute per-thread staging source pointers (advance by KC per chunk).
    const int c4 = tid & 3;
    const unsigned short* pA0 = A + (size_t)(tile_m + ((tid + 0 * 256) >> 2)) * FAN1 + c4 * 8;
    const unsigned short* pA1 = A + (size_t)(tile_m + ((tid + 1 * 256) >> 2)) * FAN1 + c4 * 8;
    const unsigned short* pA2 = A + (size_t)(tile_m + ((tid + 2 * 256) >> 2)) * FAN1 + c4 * 8;
    const unsigned short* pA3 = A + (size_t)(tile_m + ((tid + 3 * 256) >> 2)) * FAN1 + c4 * 8;
    const unsigned short* pB  = Bw + (size_t)(tile_n + (tid >> 2)) * FAN1 + c4 * 8;

    auto stage = [&](int buf, int k0) {
        copy16_g2l(&sA[buf][(tid + 0 * 256) >> 2][c4 * 8], pA0 + k0);
        copy16_g2l(&sA[buf][(tid + 1 * 256) >> 2][c4 * 8], pA1 + k0);
        copy16_g2l(&sA[buf][(tid + 2 * 256) >> 2][c4 * 8], pA2 + k0);
        copy16_g2l(&sA[buf][(tid + 3 * 256) >> 2][c4 * 8], pA3 + k0);
        copy16_g2l(&sB[buf][tid >> 2][c4 * 8], pB + k0);
    };

    stage(0, 0);
    int cur = 0;
    for (int i = 0; i < NCHUNK; ++i) {
        if (i + 1 < NCHUNK) {
            stage(cur ^ 1, (i + 1) * KC);   // prefetch next chunk (async)
            WAIT_ASYNC(5);                  // drain chunk i, keep i+1 in flight
        } else {
            WAIT_ASYNC(0);
        }
        __syncthreads();                    // chunk i visible to all waves

        v8bf a0_lo = *reinterpret_cast<const v8bf*>(&sA[cur][wrow + r16][kA]);
        v8bf a0_hi = *reinterpret_cast<const v8bf*>(&sA[cur][wrow + r16][kA + 16]);
        v8bf a1_lo = *reinterpret_cast<const v8bf*>(&sA[cur][wrow + 16 + r16][kA]);
        v8bf a1_hi = *reinterpret_cast<const v8bf*>(&sA[cur][wrow + 16 + r16][kA + 16]);
        v16bf af0 = __builtin_shufflevector(a0_lo, a0_hi,
                      0,1,2,3,4,5,6,7,8,9,10,11,12,13,14,15);
        v16bf af1 = __builtin_shufflevector(a1_lo, a1_hi,
                      0,1,2,3,4,5,6,7,8,9,10,11,12,13,14,15);

        #define BFRAG(t)                                                           \
            __builtin_shufflevector(                                               \
                *reinterpret_cast<const v8bf*>(&sB[cur][(t) * 16 + r16][kB]),      \
                *reinterpret_cast<const v8bf*>(&sB[cur][(t) * 16 + r16][kB + 16]), \
                0,1,2,3,4,5,6,7,8,9,10,11,12,13,14,15)

        {
            v16bf bf = BFRAG(0);
            c00 = __builtin_amdgcn_wmma_f32_16x16x32_bf16(false, af0, false, bf, (short)0, c00, false, false);
            c10 = __builtin_amdgcn_wmma_f32_16x16x32_bf16(false, af1, false, bf, (short)0, c10, false, false);
        }
        {
            v16bf bf = BFRAG(1);
            c01 = __builtin_amdgcn_wmma_f32_16x16x32_bf16(false, af0, false, bf, (short)0, c01, false, false);
            c11 = __builtin_amdgcn_wmma_f32_16x16x32_bf16(false, af1, false, bf, (short)0, c11, false, false);
        }
        {
            v16bf bf = BFRAG(2);
            c02 = __builtin_amdgcn_wmma_f32_16x16x32_bf16(false, af0, false, bf, (short)0, c02, false, false);
            c12 = __builtin_amdgcn_wmma_f32_16x16x32_bf16(false, af1, false, bf, (short)0, c12, false, false);
        }
        {
            v16bf bf = BFRAG(3);
            c03 = __builtin_amdgcn_wmma_f32_16x16x32_bf16(false, af0, false, bf, (short)0, c03, false, false);
            c13 = __builtin_amdgcn_wmma_f32_16x16x32_bf16(false, af1, false, bf, (short)0, c13, false, false);
        }
        #undef BFRAG

        __syncthreads();                    // all reads of buf `cur` done
        cur ^= 1;
    }

    // Epilogue: bias + relu + store.
    // C/D layout: VGPR i -> M = i + 8*hi, N = lane%16.
    auto store_tile = [&](const v8f& c, int rblk, int t) {
        const int rbase = tile_m + wrow + rblk * 16 + hi * 8;
        const int col   = tile_n + t * 16 + r16;
        const float bv  = bias[col];
        #pragma unroll
        for (int i = 0; i < 8; ++i) {
            float v = c[i] + bv;
            v = fmaxf(v, 0.f);
            H[(size_t)(rbase + i) * H1 + col] = v;
        }
    };
    store_tile(c00, 0, 0); store_tile(c01, 0, 1); store_tile(c02, 0, 2); store_tile(c03, 0, 3);
    store_tile(c10, 1, 0); store_tile(c11, 1, 1); store_tile(c12, 1, 2); store_tile(c13, 1, 3);
}

// ---------------------------------------------------------------------------
// Kernel 5: out = H @ W2^T + b2.  4096x4 outputs, L2-resident operands.
// ---------------------------------------------------------------------------
__global__ __launch_bounds__(256) void gemm2_kernel(
    const float* __restrict__ H,
    const float* __restrict__ W2,
    const float* __restrict__ b2,
    float* __restrict__ out)
{
    const int t = blockIdx.x * 256 + threadIdx.x;   // 0 .. B*NCLS-1
    if (t >= B * NCLS) return;
    const int b = t >> 2;
    const int c = t & 3;
    const float4* h4 = reinterpret_cast<const float4*>(H + (size_t)b * H1);
    const float4* w4 = reinterpret_cast<const float4*>(W2 + (size_t)c * H1);
    float acc = b2[c];
    #pragma unroll 4
    for (int i = 0; i < H1 / 4; ++i) {
        float4 x = h4[i], y = w4[i];
        acc += x.x * y.x + x.y * y.y + x.z * y.z + x.w * y.w;
    }
    out[t] = acc;
}

// ---------------------------------------------------------------------------
// Launch
// ---------------------------------------------------------------------------
extern "C" void kernel_launch(void* const* d_in, const int* in_sizes, int n_in,
                              void* d_out, int out_size, void* d_ws, size_t ws_size,
                              hipStream_t stream) {
    const float* sim_stance = (const float*)d_in[0];
    const float* nli_stance = (const float*)d_in[1];
    const float* sim_body   = (const float*)d_in[2];
    const float* nli_body   = (const float*)d_in[3];
    const float* W1         = (const float*)d_in[4];
    const float* b1         = (const float*)d_in[5];
    const float* W2         = (const float*)d_in[6];
    const float* b2         = (const float*)d_in[7];
    float* out = (float*)d_out;

    char* ws = (char*)d_ws;
    size_t off = 0;
    int* topidx = (int*)(ws + off);                    off += 128 * 1024;
    unsigned short* xx  = (unsigned short*)(ws + off); off += (size_t)B * FAN1 * 2;
    unsigned short* w1b = (unsigned short*)(ws + off); off += (size_t)H1 * FAN1 * 2;
    float* Hbuf = (float*)(ws + off);

    sims_topk_kernel<<<B, 256, 0, stream>>>(sim_stance, sim_body, topidx);
    build_xx_kernel<<<B, 256, 0, stream>>>(nli_stance, nli_body, topidx, xx);
    cvt_w1_kernel<<<2048, 256, 0, stream>>>(W1, w1b, H1 * FAN1);

    dim3 g1(B / BM, H1 / BN);   // 16 x 16
    gemm1_kernel<<<g1, 256, 0, stream>>>(xx, w1b, b1, Hbuf);

    gemm2_kernel<<<(B * NCLS + 255) / 256, 256, 0, stream>>>(Hbuf, W2, b2, out);
}
